// NonEquivariantAttention_69741678952450
// MI455X (gfx1250) — matrix operand order repaired and checked by reading, not compiled, not run on hardware
//
#include <hip/hip_runtime.h>
#include <hip/hip_bf16.h>
#include <stdint.h>

typedef __attribute__((ext_vector_type(16))) __bf16 v16bf;
typedef __attribute__((ext_vector_type(8)))  float  v8f;

#define N_NODES 30000
#define N_EDGES 480000
#define HD      512
#define HID     256
#define EA_K    288   // 263 padded to 9 K-tiles of 32

// ---------------------------------------------------------------- helpers

__device__ __forceinline__ float silu_f(float v) { return v / (1.f + __expf(-v)); }

// monotone float <-> uint encoding for atomic segment-max
__device__ __forceinline__ unsigned enc_f32(float f) {
    unsigned b = __float_as_uint(f);
    return (b & 0x80000000u) ? ~b : (b | 0x80000000u);
}
__device__ __forceinline__ float dec_f32(unsigned u) {
    unsigned b = (u & 0x80000000u) ? (u ^ 0x80000000u) : ~u;
    return __uint_as_float(b);
}

// A fragment (16x32 bf16, MxK): lane&15 = M, K = (e<8?e:e+8) + 8*(lane>=16)
__device__ __forceinline__ v16bf load_a(const __bf16* p, int ld, int rbase, int kbase, int lane) {
    int r  = rbase + (lane & 15);
    int kb = kbase + ((lane & 16) >> 1);      // +8 for upper half-wave
    const __bf16* q = p + (size_t)r * ld + kb;
    union { v16bf v; uint4 x[2]; } u;
    u.x[0] = *reinterpret_cast<const uint4*>(q);
    u.x[1] = *reinterpret_cast<const uint4*>(q + 16);
    return u.v;
}

// B fragment (32x16 bf16, KxN) loaded from W stored row-major [N][K]:
// lane&15 = N, K = (e<8?e:e+8) + 8*(lane>=16)
__device__ __forceinline__ v16bf load_b(const __bf16* p, int ld, int nbase, int kbase, int lane) {
    int r  = nbase + (lane & 15);
    int kb = kbase + ((lane & 16) >> 1);      // +8 for upper half-wave
    const __bf16* q = p + (size_t)r * ld + kb;
    union { v16bf v; uint4 x[2]; } u;
    u.x[0] = *reinterpret_cast<const uint4*>(q);
    u.x[1] = *reinterpret_cast<const uint4*>(q + 16);
    return u.v;
}

// A fragment sourced from fp32 (converted on the fly)
__device__ __forceinline__ v16bf load_a_f32(const float* p, int ld, int rbase, int kbase, int lane) {
    int r  = rbase + (lane & 15);
    int kb = kbase + ((lane & 16) >> 1);
    const float* q = p + (size_t)r * ld + kb;
    v16bf v;
#pragma unroll
    for (int i = 0; i < 8; ++i) v[i] = (__bf16)q[i];
#pragma unroll
    for (int i = 0; i < 8; ++i) v[8 + i] = (__bf16)q[16 + i];
    return v;
}

__device__ __forceinline__ v8f wmma_bf16(v16bf a, v16bf b, v8f c) {
    return __builtin_amdgcn_wmma_f32_16x16x32_bf16(false, a, false, b, (short)0, c, false, false);
}

// ---------------------------------------------------------------- small kernels

__global__ void k_init(unsigned* smax, float* ssum, float* agg) {
    size_t stride = (size_t)gridDim.x * blockDim.x;
    for (size_t i = (size_t)blockIdx.x * blockDim.x + threadIdx.x;
         i < (size_t)N_NODES * HD; i += stride) {
        agg[i] = 0.f;
        if (i < (size_t)N_NODES * 8) { smax[i] = enc_f32(-3.402823e38f); ssum[i] = 0.f; }
    }
}

__global__ void k_cvt(const float* src, __bf16* dst, int n) {
    int stride = gridDim.x * blockDim.x;
    for (int i = blockIdx.x * blockDim.x + threadIdx.x; i < n; i += stride)
        dst[i] = (__bf16)src[i];
}

__global__ void k_cvt_pad(const float* src, __bf16* dst, int srows, int scols, int drows, int dcols) {
    int n = drows * dcols, stride = gridDim.x * blockDim.x;
    for (int i = blockIdx.x * blockDim.x + threadIdx.x; i < n; i += stride) {
        int r = i / dcols, c = i % dcols;
        dst[i] = (r < srows && c < scols) ? (__bf16)src[r * scols + c] : (__bf16)0.f;
    }
}

__global__ void k_zero_tail(float* p, int n) {
    int stride = gridDim.x * blockDim.x;
    for (int i = blockIdx.x * blockDim.x + threadIdx.x; i < n; i += stride) p[i] = 0.f;
}

// ---------------------------------------------------------------- q/k projection (bf16 result)
// grid.x = 1875, block = 256 (8 waves). [16 nodes] x [512 cols], K=128.
__global__ void k_proj_qk(const __bf16* A, const __bf16* W, const float* bias, __bf16* outb) {
    int lane = threadIdx.x & 31, wave = threadIdx.x >> 5;
    int mbase = blockIdx.x * 16;
    for (int nt = wave; nt < 32; nt += 8) {
        v8f acc = {};
#pragma unroll
        for (int kt = 0; kt < 4; ++kt) {
            v16bf a = load_a(A, 128, mbase, kt * 32, lane);
            v16bf b = load_b(W, 128, nt * 16, kt * 32, lane);
            acc = wmma_bf16(a, b, acc);
        }
        int col = nt * 16 + (lane & 15);
        float bv = bias[col];
#pragma unroll
        for (int vi = 0; vi < 8; ++vi) {
            int m = mbase + vi + ((lane & 16) >> 1);
            outb[(size_t)m * HD + col] = (__bf16)(acc[vi] + bv);
        }
    }
}

// v projection (fp32 result, kept full precision for aggregation)
__global__ void k_proj_v(const __bf16* A, const __bf16* W, const float* bias, float* outf) {
    int lane = threadIdx.x & 31, wave = threadIdx.x >> 5;
    int mbase = blockIdx.x * 16;
    for (int nt = wave; nt < 32; nt += 8) {
        v8f acc = {};
#pragma unroll
        for (int kt = 0; kt < 4; ++kt) {
            v16bf a = load_a(A, 128, mbase, kt * 32, lane);
            v16bf b = load_b(W, 128, nt * 16, kt * 32, lane);
            acc = wmma_bf16(a, b, acc);
        }
        int col = nt * 16 + (lane & 15);
        float bv = bias[col];
#pragma unroll
        for (int vi = 0; vi < 8; ++vi) {
            int m = mbase + vi + ((lane & 16) >> 1);
            outf[(size_t)m * HD + col] = acc[vi] + bv;
        }
    }
}

// ---------------------------------------------------------------- fused edge kernel
// One block per 16-edge tile: async global->LDS gather of h[row]|h[col] ->
// MLP L1 (WMMA, silu) -> L2 (WMMA) -> eww head-weights (WMMA, wave0)
// + pos-enc (VALU) + q.k score + mask + atomic segment-max.
__global__ void k_edge(const __bf16* hbf, const float* x, const int* ei, const unsigned char* mask,
                       const __bf16* ew1p, const float* eb1,
                       const __bf16* ew2b, const float* eb2,
                       const __bf16* ewwp, const float* ewb,
                       const float* pw1, const float* pb1,
                       const float* pw2, const float* pb2,
                       const __bf16* qbf, const __bf16* kbf,
                       float* score, unsigned* smax) {
    __shared__ __align__(16) __bf16 sAttr[16][EA_K];
    __shared__ __align__(16) __bf16 sH1[16][HID];
    __shared__ __align__(16) __bf16 sEF[16][HID];
    __shared__ float sPosH[16][HID];
    __shared__ float sPos[16][8];
    __shared__ float sEw[16][8];
    __shared__ int   sRow[16], sCol[16];
    __shared__ float sX[16][7];   // [0]=rel_dist, [1..3]=xr, [4..6]=xc

    int tid = threadIdx.x, lane = tid & 31, wave = tid >> 5;
    int ebase = blockIdx.x * 16;

    if (tid < 16) {
        int r = ei[ebase + tid];
        int c = ei[N_EDGES + ebase + tid];
        sRow[tid] = r; sCol[tid] = c;
        float xr0 = x[r * 3], xr1 = x[r * 3 + 1], xr2 = x[r * 3 + 2];
        float xc0 = x[c * 3], xc1 = x[c * 3 + 1], xc2 = x[c * 3 + 2];
        float d0 = xr0 - xc0, d1 = xr1 - xc1, d2 = xr2 - xc2;
        sX[tid][0] = d0 * d0 + d1 * d1 + d2 * d2;
        sX[tid][1] = xr0; sX[tid][2] = xr1; sX[tid][3] = xr2;
        sX[tid][4] = xc0; sX[tid][5] = xc1; sX[tid][6] = xc2;
    }
    __syncthreads();

    // async global->LDS: h[row] into cols [0,128), h[col] into cols [128,256)
    // 16 edges x 32 chunks of 16B; LDS dest = low 32 bits of generic shared ptr
    unsigned ldsBase = (unsigned)(uintptr_t)&sAttr[0][0];
    for (int j = tid; j < 16 * 32; j += 256) {
        int e = j >> 5, ch = j & 31;
        const __bf16* src = (ch < 16) ? (hbf + (size_t)sRow[e] * 128 + ch * 8)
                                      : (hbf + (size_t)sCol[e] * 128 + (ch - 16) * 8);
        unsigned dst = ldsBase + (unsigned)(e * EA_K + ch * 8) * 2u;
        asm volatile("global_load_async_to_lds_b128 %0, %1, off"
                     :: "v"(dst), "v"(src) : "memory");
    }
    // fill geometric columns [256,288) with rel|xr|xc|zero-pad (plain LDS stores)
    for (int j = tid; j < 16 * 32; j += 256) {
        int e = j >> 5, c = 256 + (j & 31);
        sAttr[e][c] = (c < 263) ? (__bf16)sX[e][c - 256] : (__bf16)0.f;
    }
    asm volatile("s_wait_asynccnt 0x0" ::: "memory");
    __syncthreads();

    // layer 1: hidden1 = silu(attr @ ew1^T + eb1); each wave -> 32 columns
    {
        v8f a0 = {}, a1 = {};
#pragma unroll
        for (int kt = 0; kt < 9; ++kt) {
            v16bf a  = load_a(&sAttr[0][0], EA_K, 0, kt * 32, lane);
            v16bf b0 = load_b(ew1p, EA_K, wave * 32, kt * 32, lane);
            v16bf b1 = load_b(ew1p, EA_K, wave * 32 + 16, kt * 32, lane);
            a0 = wmma_bf16(a, b0, a0);
            a1 = wmma_bf16(a, b1, a1);
        }
        int c0 = wave * 32 + (lane & 15), c1 = c0 + 16;
        float bb0 = eb1[c0], bb1 = eb1[c1];
#pragma unroll
        for (int vi = 0; vi < 8; ++vi) {
            int m = vi + ((lane & 16) >> 1);
            sH1[m][c0] = (__bf16)silu_f(a0[vi] + bb0);
            sH1[m][c1] = (__bf16)silu_f(a1[vi] + bb1);
        }
    }
    __syncthreads();

    // layer 2: edge_feat = h1 @ ew2^T + eb2 (no activation)
    {
        v8f a0 = {}, a1 = {};
#pragma unroll
        for (int kt = 0; kt < 8; ++kt) {
            v16bf a  = load_a(&sH1[0][0], HID, 0, kt * 32, lane);
            v16bf b0 = load_b(ew2b, HID, wave * 32, kt * 32, lane);
            v16bf b1 = load_b(ew2b, HID, wave * 32 + 16, kt * 32, lane);
            a0 = wmma_bf16(a, b0, a0);
            a1 = wmma_bf16(a, b1, a1);
        }
        int c0 = wave * 32 + (lane & 15), c1 = c0 + 16;
        float bb0 = eb2[c0], bb1 = eb2[c1];
#pragma unroll
        for (int vi = 0; vi < 8; ++vi) {
            int m = vi + ((lane & 16) >> 1);
            sEF[m][c0] = (__bf16)(a0[vi] + bb0);
            sEF[m][c1] = (__bf16)(a1[vi] + bb1);
        }
    }
    __syncthreads();

    // pos-enc hidden (all threads, VALU) while wave0 runs the tiny eww WMMA
    for (int i = tid; i < 16 * HID; i += 256) {
        int e = i >> 8, hc = i & 255;
        const float* w = pw1 + hc * 6;
        float s = pb1[hc] + w[0] * sX[e][1] + w[1] * sX[e][2] + w[2] * sX[e][3]
                          + w[3] * sX[e][4] + w[4] * sX[e][5] + w[5] * sX[e][6];
        sPosH[e][hc] = silu_f(s);
    }
    if (wave == 0) {
        v8f acc = {};
#pragma unroll
        for (int kt = 0; kt < 8; ++kt) {
            v16bf a = load_a(&sEF[0][0], HID, 0, kt * 32, lane);
            v16bf b = load_b(ewwp, HID, 0, kt * 32, lane);
            acc = wmma_bf16(a, b, acc);
        }
        int n = lane & 15;
#pragma unroll
        for (int vi = 0; vi < 8; ++vi) {
            int m = vi + ((lane & 16) >> 1);
            if (n < 8) sEw[m][n] = acc[vi] + ewb[n];
        }
    }
    __syncthreads();

    // pos-enc reduce: [16,256] @ pw2^T -> [16,8]
    if (tid < 128) {
        int e = tid >> 3, hh = tid & 7;
        const float* w = pw2 + hh * 256;
        float s = pb2[hh];
        for (int j = 0; j < 256; ++j) s += sPosH[e][j] * w[j];
        sPos[e][hh] = s;
    }
    __syncthreads();

    // final score: q.k/sqrt(64) + pos + edge_w, mask, segment-max
    if (tid < 128) {
        int e = tid >> 3, hh = tid & 7;
        int ge = ebase + e;
        int r = sRow[e], c = sCol[e];
        const __bf16* qp = qbf + (size_t)r * HD + hh * 64;
        const __bf16* kp = kbf + (size_t)c * HD + hh * 64;
        float s = 0.f;
#pragma unroll 8
        for (int d = 0; d < 64; ++d) s += (float)qp[d] * (float)kp[d];
        s = s * 0.125f + sPos[e][hh] + sEw[e][hh];
        if (!mask[ge]) s = -1e9f;
        score[(size_t)ge * 8 + hh] = s;
        atomicMax(&smax[(size_t)r * 8 + hh], enc_f32(s));
    }
}

// ---------------------------------------------------------------- softmax passes + aggregation

__global__ void k_escore(const int* ei, float* score, const unsigned* smax, float* ssum) {
    int stride = gridDim.x * blockDim.x;
    for (int i = blockIdx.x * blockDim.x + threadIdx.x; i < N_EDGES * 8; i += stride) {
        int e = i >> 3, hh = i & 7;
        int r = ei[e];
        float es = __expf(score[i] - dec_f32(smax[r * 8 + hh]));
        score[i] = es;
        atomicAdd(&ssum[r * 8 + hh], es);
    }
}

__global__ void k_norm(const int* ei, float* score, const float* ssum) {
    int stride = gridDim.x * blockDim.x;
    for (int i = blockIdx.x * blockDim.x + threadIdx.x; i < N_EDGES * 8; i += stride) {
        int e = i >> 3, hh = i & 7;
        score[i] = score[i] / (ssum[ei[e] * 8 + hh] + 1e-8f);
    }
}

__global__ void k_agg(const int* ei, const float* attn, const float* vf, float* agg) {
    size_t total = (size_t)N_EDGES * HD;
    size_t stride = (size_t)gridDim.x * blockDim.x;
    for (size_t i = (size_t)blockIdx.x * blockDim.x + threadIdx.x; i < total; i += stride) {
        size_t e = i >> 9; int d = (int)(i & 511); int hh = d >> 6;
        int r = ei[e], c = ei[N_EDGES + e];
        float w = attn[e * 8 + hh];
        atomicAdd(&agg[(size_t)r * HD + d], w * vf[(size_t)c * HD + d]);
    }
}

// ---------------------------------------------------------------- coord MLP: 3 -> 256 -> 256 -> 128
__global__ void k_coord(const float* x, const float* cw1, const float* cb1,
                        const __bf16* cw2b, const float* cb2,
                        const __bf16* cw3b, const float* cb3, float* coord) {
    __shared__ __align__(16) __bf16 sC1[16][HID];
    __shared__ __align__(16) __bf16 sC2[16][HID];
    int tid = threadIdx.x, lane = tid & 31, wave = tid >> 5;
    int nbase = blockIdx.x * 16;

    for (int i = tid; i < 16 * HID; i += 256) {
        int m = i >> 8, hc = i & 255;
        int node = nbase + m;
        const float* w = cw1 + hc * 3;
        float s = cb1[hc] + w[0] * x[node * 3] + w[1] * x[node * 3 + 1] + w[2] * x[node * 3 + 2];
        sC1[m][hc] = (__bf16)silu_f(s);
    }
    __syncthreads();

    {   // layer2 + silu
        v8f a0 = {}, a1 = {};
#pragma unroll
        for (int kt = 0; kt < 8; ++kt) {
            v16bf a  = load_a(&sC1[0][0], HID, 0, kt * 32, lane);
            v16bf b0 = load_b(cw2b, HID, wave * 32, kt * 32, lane);
            v16bf b1 = load_b(cw2b, HID, wave * 32 + 16, kt * 32, lane);
            a0 = wmma_bf16(a, b0, a0);
            a1 = wmma_bf16(a, b1, a1);
        }
        int c0 = wave * 32 + (lane & 15), c1 = c0 + 16;
        float bb0 = cb2[c0], bb1 = cb2[c1];
#pragma unroll
        for (int vi = 0; vi < 8; ++vi) {
            int m = vi + ((lane & 16) >> 1);
            sC2[m][c0] = (__bf16)silu_f(a0[vi] + bb0);
            sC2[m][c1] = (__bf16)silu_f(a1[vi] + bb1);
        }
    }
    __syncthreads();

    {   // layer3 -> coord [n,128]
        v8f acc = {};
#pragma unroll
        for (int kt = 0; kt < 8; ++kt) {
            v16bf a = load_a(&sC2[0][0], HID, 0, kt * 32, lane);
            v16bf b = load_b(cw3b, HID, wave * 16, kt * 32, lane);
            acc = wmma_bf16(a, b, acc);
        }
        int col = wave * 16 + (lane & 15);
        float bb = cb3[col];
#pragma unroll
        for (int vi = 0; vi < 8; ++vi) {
            int m = nbase + vi + ((lane & 16) >> 1);
            coord[(size_t)m * 128 + col] = acc[vi] + bb;
        }
    }
}

// ---------------------------------------------------------------- out = agg @ wo^T + bo + coord
__global__ void k_out(const float* agg, const __bf16* wob, const float* bo,
                      const float* coord, float* out) {
    int lane = threadIdx.x & 31, wave = threadIdx.x >> 5;
    int nbase = blockIdx.x * 16;
    v8f acc = {};
#pragma unroll
    for (int kt = 0; kt < 16; ++kt) {
        v16bf a = load_a_f32(agg, HD, nbase, kt * 32, lane);
        v16bf b = load_b(wob, HD, wave * 16, kt * 32, lane);
        acc = wmma_bf16(a, b, acc);
    }
    int col = wave * 16 + (lane & 15);
    float bb = bo[col];
#pragma unroll
    for (int vi = 0; vi < 8; ++vi) {
        int m = nbase + vi + ((lane & 16) >> 1);
        out[(size_t)m * 128 + col] = acc[vi] + bb + coord[(size_t)m * 128 + col];
    }
}

// ---------------------------------------------------------------- launch

extern "C" void kernel_launch(void* const* d_in, const int* in_sizes, int n_in,
                              void* d_out, int out_size, void* d_ws, size_t ws_size,
                              hipStream_t stream) {
    (void)in_sizes; (void)n_in; (void)out_size; (void)ws_size;
    const float* h   = (const float*)d_in[0];
    const float* x   = (const float*)d_in[1];
    const int*   ei  = (const int*)d_in[2];                 // edge_index (x64 off -> int32)
    const unsigned char* mask = (const unsigned char*)d_in[3];
    const float* wq = (const float*)d_in[4];  const float* bq = (const float*)d_in[5];
    const float* wk = (const float*)d_in[6];  const float* bk = (const float*)d_in[7];
    const float* wv = (const float*)d_in[8];  const float* bv = (const float*)d_in[9];
    const float* wo = (const float*)d_in[10]; const float* bo = (const float*)d_in[11];
    const float* pw1 = (const float*)d_in[12]; const float* pb1 = (const float*)d_in[13];
    const float* pw2 = (const float*)d_in[14]; const float* pb2 = (const float*)d_in[15];
    const float* ew1 = (const float*)d_in[16]; const float* eb1 = (const float*)d_in[17];
    const float* ew2 = (const float*)d_in[18]; const float* eb2 = (const float*)d_in[19];
    const float* eww = (const float*)d_in[20]; const float* ewb = (const float*)d_in[21];
    const float* cw1 = (const float*)d_in[22]; const float* cb1 = (const float*)d_in[23];
    const float* cw2 = (const float*)d_in[24]; const float* cb2 = (const float*)d_in[25];
    const float* cw3 = (const float*)d_in[26]; const float* cb3 = (const float*)d_in[27];
    float* out = (float*)d_out;

    char* ws = (char*)d_ws;
    size_t o = 0;
    auto alloc = [&](size_t bytes) { size_t r = o; o += (bytes + 255) & ~(size_t)255; return r; };
    __bf16* HBF  = (__bf16*)(ws + alloc((size_t)N_NODES * 128 * 2));
    __bf16* QBF  = (__bf16*)(ws + alloc((size_t)N_NODES * HD * 2));
    __bf16* KBF  = (__bf16*)(ws + alloc((size_t)N_NODES * HD * 2));
    float*  VF   = (float*) (ws + alloc((size_t)N_NODES * HD * 4));
    float*  SC   = (float*) (ws + alloc((size_t)N_EDGES * 8 * 4));
    unsigned* SMAX = (unsigned*)(ws + alloc((size_t)N_NODES * 8 * 4));
    float*  SSUM = (float*) (ws + alloc((size_t)N_NODES * 8 * 4));
    float*  AGG  = (float*) (ws + alloc((size_t)N_NODES * HD * 4));
    float*  CRD  = (float*) (ws + alloc((size_t)N_NODES * 128 * 4));
    __bf16* WQB  = (__bf16*)(ws + alloc((size_t)HD * 128 * 2));
    __bf16* WKB  = (__bf16*)(ws + alloc((size_t)HD * 128 * 2));
    __bf16* WVB  = (__bf16*)(ws + alloc((size_t)HD * 128 * 2));
    __bf16* WOB  = (__bf16*)(ws + alloc((size_t)128 * HD * 2));
    __bf16* EW1P = (__bf16*)(ws + alloc((size_t)HID * EA_K * 2));
    __bf16* EW2B = (__bf16*)(ws + alloc((size_t)HID * HID * 2));
    __bf16* EWWP = (__bf16*)(ws + alloc((size_t)16 * HID * 2));
    __bf16* CW2B = (__bf16*)(ws + alloc((size_t)HID * HID * 2));
    __bf16* CW3B = (__bf16*)(ws + alloc((size_t)128 * HID * 2));

    k_init<<<4096, 256, 0, stream>>>(SMAX, SSUM, AGG);

    k_cvt<<<4096, 256, 0, stream>>>(h,   HBF,  N_NODES * 128);
    k_cvt<<<256,  256, 0, stream>>>(wq,  WQB,  HD * 128);
    k_cvt<<<256,  256, 0, stream>>>(wk,  WKB,  HD * 128);
    k_cvt<<<256,  256, 0, stream>>>(wv,  WVB,  HD * 128);
    k_cvt<<<256,  256, 0, stream>>>(wo,  WOB,  128 * HD);
    k_cvt<<<256,  256, 0, stream>>>(ew2, EW2B, HID * HID);
    k_cvt<<<256,  256, 0, stream>>>(cw2, CW2B, HID * HID);
    k_cvt<<<128,  256, 0, stream>>>(cw3, CW3B, 128 * HID);
    k_cvt_pad<<<288, 256, 0, stream>>>(ew1, EW1P, HID, 263, HID, EA_K);
    k_cvt_pad<<<16,  256, 0, stream>>>(eww, EWWP, 8, HID, 16, HID);

    k_proj_qk<<<N_NODES / 16, 256, 0, stream>>>(HBF, WQB, bq, QBF);
    k_proj_qk<<<N_NODES / 16, 256, 0, stream>>>(HBF, WKB, bk, KBF);
    k_proj_v <<<N_NODES / 16, 256, 0, stream>>>(HBF, WVB, bv, VF);

    k_coord<<<N_NODES / 16, 256, 0, stream>>>(x, cw1, cb1, CW2B, cb2, CW3B, cb3, CRD);

    k_edge<<<N_EDGES / 16, 256, 0, stream>>>(HBF, x, ei, mask,
                                             EW1P, eb1, EW2B, eb2, EWWP, ewb,
                                             pw1, pb1, pw2, pb2,
                                             QBF, KBF, SC, SMAX);

    k_escore<<<4096, 256, 0, stream>>>(ei, SC, SMAX, SSUM);
    k_norm<<<4096, 256, 0, stream>>>(ei, SC, SSUM);
    k_agg<<<32768, 256, 0, stream>>>(ei, SC, VF, AGG);

    k_out<<<N_NODES / 16, 256, 0, stream>>>(AGG, WOB, bo, CRD, out);
    k_zero_tail<<<352, 256, 0, stream>>>(out + (size_t)N_NODES * 128, N_NODES * 3);
}
